// FakeNewsDetector_18545668784934
// MI455X (gfx1250) — matrix operand order repaired and checked by reading, not compiled, not run on hardware
//
#include <hip/hip_runtime.h>

// ---------------------------------------------------------------------------
// GCN (3x GCNConv + LN + ReLU, mean-pool, MLP) for MI455X / gfx1250.
// N=100000 nodes, E=1600000 edges, D=H=128, B=64.
//
// Key MI455X reasoning:
//  * h (100k x 128 f32 = 51.2MB) fits in the 192MB L2 -> edge gathers and
//    scatter atomics are L2 traffic, not HBM. Workload is memory/atomic bound.
//  * GEMM is ~10 GFLOP total -> done with v_wmma_f32_16x16x32_bf16 (bf16 in,
//    f32 accumulate). W is pre-swizzled to B-operand layout & staged in LDS.
//  * Symmetric normalization dis[s]*dis[d] folded into GEMM epilogue (dis[s])
//    and LN prologue (dis[d]) to avoid extra passes.
// ---------------------------------------------------------------------------

#define GCN_N 100000
#define GCN_E 1600000
#define GCN_H 128
#define GCN_B 64

typedef __attribute__((ext_vector_type(16))) __bf16 v16bf;
typedef __attribute__((ext_vector_type(8)))  __bf16 v8bf;
typedef __attribute__((ext_vector_type(4)))  __bf16 v4bf;
typedef __attribute__((ext_vector_type(8)))  float  v8f;

__device__ __forceinline__ __bf16 f2bf(float f) {
    unsigned u = __builtin_bit_cast(unsigned, f);
    u += 0x7fffu + ((u >> 16) & 1u);            // round-to-nearest-even
    unsigned short h = (unsigned short)(u >> 16);
    return __builtin_bit_cast(__bf16, h);
}

// ---------------------------------------------------------------- degree/dis
__global__ void k_dis_init(float* __restrict__ dis, int n) {
    int i = blockIdx.x * blockDim.x + threadIdx.x;
    if (i < n) dis[i] = 1.0f;                   // self-loop contributes 1
}

__global__ void k_deg_edges(float* __restrict__ dis,
                            const long long* __restrict__ ei, int e) {
    int i = blockIdx.x * blockDim.x + threadIdx.x;
    if (i < e) atomicAdd(&dis[(int)ei[(long long)e + i]], 1.0f); // dst side
}

__global__ void k_dis_rsqrt(float* __restrict__ dis, int n) {
    int i = blockIdx.x * blockDim.x + threadIdx.x;
    if (i < n) dis[i] = rsqrtf(dis[i]);
}

// ------------------------------------------------------------ f32 -> bf16 cvt
__global__ void k_to_bf16(const float* __restrict__ x, __bf16* __restrict__ hb,
                          long long cnt) {
    long long i = (long long)blockIdx.x * blockDim.x + threadIdx.x;
    if (i < cnt) hb[i] = f2bf(x[i]);
}

// Swizzle W (row-major [K=128, N=128] f32) into this code's contiguous-K B-operand lane layout
// (16-bit B 32x16: lanes 0-15 hold col=lane K=0..15, lanes 16-31 col=lane-16
// K=16..31, 2 elems per VGPR). Fragment index = (ntile*4 + kchunk).
__global__ void k_convW(const float* __restrict__ W, __bf16* __restrict__ Wb) {
    int i = blockIdx.x * blockDim.x + threadIdx.x;
    if (i >= GCN_H * GCN_H) return;
    int j    = i & 15;          // element within lane (contiguous K)
    int l    = (i >> 4) & 31;   // lane
    int frag = i >> 9;          // 0..31  (= ntile*4 + kc)
    int kc   = frag & 3;
    int nt   = frag >> 2;
    int col  = nt * 16 + (l & 15);
    int k    = kc * 32 + (l >> 4) * 16 + j;
    Wb[i] = f2bf(W[k * GCN_H + col]);
}

// ---------------------------------------------------------------- WMMA GEMM
// t[row,:] = (hb[row,:] @ W) * dis[row].  One wave -> 16x128 output strip.
__global__ void __launch_bounds__(256)
k_gemm_wmma(const __bf16* __restrict__ hb, const __bf16* __restrict__ Wb,
            const float* __restrict__ dis, float* __restrict__ tout, int n) {
    __shared__ __bf16 sW[GCN_H * GCN_H];        // 32 KB, whole W in B-layout
    {
        const uint4* s = (const uint4*)Wb;
        uint4*       d = (uint4*)sW;
        for (int i = threadIdx.x; i < GCN_H * GCN_H / 8; i += blockDim.x)
            d[i] = s[i];
    }
    __syncthreads();

    const int wave = threadIdx.x >> 5;
    const int lane = threadIdx.x & 31;
    const int half = lane >> 4;
    const int lp   = lane & 15;
    const int m0   = blockIdx.x * 128 + wave * 16;

    v8f acc[8] = {};                            // 8 N-tiles of 16x16 f32

#pragma unroll
    for (int kc = 0; kc < 4; ++kc) {
        // A fragment, contiguous K order (matches Wb):
        //  lanes 0-15 : M=lane, elems 0-15 -> K=kc*32+0..15
        //  lanes16-31 : M=lane-16, elems 0-15 -> K=kc*32+16..31
        int row = m0 + lp;
        int rl  = row < n ? row : n - 1;        // clamp (stores are guarded)
        const __bf16* ap = hb + (long long)rl * GCN_H + kc * 32 + half * 16;
        v8bf lo = *(const v8bf*)(ap);
        v8bf hi = *(const v8bf*)(ap + 8);
        v16bf a;
#pragma unroll
        for (int j = 0; j < 8; ++j) { a[j] = lo[j]; a[j + 8] = hi[j]; }

#pragma unroll
        for (int t = 0; t < 8; ++t) {
            v16bf b = *(const v16bf*)(sW + ((t * 4 + kc) * 32 + lane) * 16);
            acc[t] = __builtin_amdgcn_wmma_f32_16x16x32_bf16(
                false, a, false, b, (short)0, acc[t], false, false);
        }
    }

    // Epilogue: C layout lanes0-15 N=lane M=v, lanes16-31 N=lane-16 M=v+8.
#pragma unroll
    for (int v = 0; v < 8; ++v) {
        int row = m0 + v + half * 8;
        if (row < n) {
            float s = dis[row];
#pragma unroll
            for (int t = 0; t < 8; ++t)
                tout[(long long)row * GCN_H + t * 16 + lp] = acc[t][v] * s;
        }
    }
}

// -------------------------------------------------------- self-loop + scatter
__global__ void k_copy(const float* __restrict__ src, float* __restrict__ dst,
                       long long cnt) {
    long long i = (long long)blockIdx.x * blockDim.x + threadIdx.x;
    if (i < cnt) dst[i] = src[i];
}

// one wave32 per edge; lane handles 4 features; atomics land in L2.
__global__ void __launch_bounds__(256)
k_scatter(const float* __restrict__ t, float* __restrict__ acc,
          const long long* __restrict__ ei, long long e) {
    const int lane = threadIdx.x & 31;
    long long w  = (long long)blockIdx.x * (blockDim.x >> 5) + (threadIdx.x >> 5);
    long long nw = (long long)gridDim.x * (blockDim.x >> 5);
    for (long long k = w; k < e; k += nw) {
        __builtin_prefetch(&ei[k + nw], 0, 1);           // global_prefetch_b8
        long long s = ei[k];
        long long d = ei[e + k];
        const float4 v = *(const float4*)(t + s * GCN_H + lane * 4);
        float* p = acc + d * GCN_H + lane * 4;
        atomicAdd(p + 0, v.x);
        atomicAdd(p + 1, v.y);
        atomicAdd(p + 2, v.z);
        atomicAdd(p + 3, v.w);
    }
}

// ------------------------------------------- dis[d]*agg + bias, LN, ReLU, cvt
__global__ void __launch_bounds__(256)
k_finalize(const float* __restrict__ acc, const float* __restrict__ dis,
           const float* __restrict__ bias, const float* __restrict__ g,
           const float* __restrict__ be, float* __restrict__ hout,
           __bf16* __restrict__ hbout, int n) {
    const int lane = threadIdx.x & 31;
    const int node = blockIdx.x * (blockDim.x >> 5) + (threadIdx.x >> 5);
    if (node >= n) return;

    float  s  = dis[node];
    float4 v  = *(const float4*)(acc + (long long)node * GCN_H + lane * 4);
    float4 bb = *(const float4*)(bias + lane * 4);
    v.x = v.x * s + bb.x; v.y = v.y * s + bb.y;
    v.z = v.z * s + bb.z; v.w = v.w * s + bb.w;

    float sum = v.x + v.y + v.z + v.w;
    float sq  = v.x * v.x + v.y * v.y + v.z * v.z + v.w * v.w;
#pragma unroll
    for (int m = 16; m >= 1; m >>= 1) {                  // wave32 reduction
        sum += __shfl_xor(sum, m, 32);
        sq  += __shfl_xor(sq,  m, 32);
    }
    float mean = sum * (1.0f / GCN_H);
    float var  = sq * (1.0f / GCN_H) - mean * mean;
    float r    = rsqrtf(var + 1e-5f);

    float4 gg = *(const float4*)(g  + lane * 4);
    float4 ee = *(const float4*)(be + lane * 4);
    float4 y;
    y.x = fmaxf((v.x - mean) * r * gg.x + ee.x, 0.0f);
    y.y = fmaxf((v.y - mean) * r * gg.y + ee.y, 0.0f);
    y.z = fmaxf((v.z - mean) * r * gg.z + ee.z, 0.0f);
    y.w = fmaxf((v.w - mean) * r * gg.w + ee.w, 0.0f);

    *(float4*)(hout + (long long)node * GCN_H + lane * 4) = y;
    v4bf o; o[0] = f2bf(y.x); o[1] = f2bf(y.y); o[2] = f2bf(y.z); o[3] = f2bf(y.w);
    *(v4bf*)(hbout + (long long)node * GCN_H + lane * 4) = o;
}

// ------------------------------------------------------------------- pooling
__global__ void k_zero(float* __restrict__ p, int cnt) {
    int i = blockIdx.x * blockDim.x + threadIdx.x;
    if (i < cnt) p[i] = 0.0f;
}

__global__ void __launch_bounds__(256)
k_pool(const float* __restrict__ h, const long long* __restrict__ batch,
       float* __restrict__ pooled, float* __restrict__ cnt, int n) {
    const int lane = threadIdx.x & 31;
    const int node = blockIdx.x * (blockDim.x >> 5) + (threadIdx.x >> 5);
    if (node >= n) return;
    long long b = batch[node];
    const float4 v = *(const float4*)(h + (long long)node * GCN_H + lane * 4);
    float* p = pooled + b * GCN_H + lane * 4;
    atomicAdd(p + 0, v.x);
    atomicAdd(p + 1, v.y);
    atomicAdd(p + 2, v.z);
    atomicAdd(p + 3, v.w);
    if (lane == 0) atomicAdd(cnt + b, 1.0f);
}

// -------------------------------------------------------------------- MLP
__global__ void k_mlp(const float* __restrict__ pooled,
                      const float* __restrict__ cnt,
                      const float* __restrict__ fw1, const float* __restrict__ fb1,
                      const float* __restrict__ fw2, const float* __restrict__ fb2,
                      float* __restrict__ out) {
    __shared__ float pm[GCN_H];
    __shared__ float z1[GCN_H / 2];
    const int b   = blockIdx.x;
    const int tid = threadIdx.x;                 // 64 threads
    float inv = 1.0f / fmaxf(cnt[b], 1.0f);
    pm[tid]      = pooled[b * GCN_H + tid] * inv;
    pm[tid + 64] = pooled[b * GCN_H + tid + 64] * inv;
    __syncthreads();

    float a = fb1[tid];
    for (int k = 0; k < GCN_H; ++k) a += pm[k] * fw1[k * 64 + tid];
    z1[tid] = fmaxf(a, 0.0f);
    __syncthreads();

    if (tid < 2) {
        float o = fb2[tid];
        for (int j = 0; j < 64; ++j) o += z1[j] * fw2[j * 2 + tid];
        out[b * 2 + tid] = o;
    }
}

// ---------------------------------------------------------------------------
static inline size_t alignup(size_t x) { return (x + 255) & ~(size_t)255; }

extern "C" void kernel_launch(void* const* d_in, const int* in_sizes, int n_in,
                              void* d_out, int out_size, void* d_ws, size_t ws_size,
                              hipStream_t stream) {
    (void)in_sizes; (void)n_in; (void)out_size; (void)ws_size;
    const int N = GCN_N, E = GCN_E, H = GCN_H, B = GCN_B;

    const float*     x     = (const float*)d_in[0];
    const long long* ei    = (const long long*)d_in[1];
    const long long* batch = (const long long*)d_in[2];
    const float* W[3]  = { (const float*)d_in[3],  (const float*)d_in[7],  (const float*)d_in[11] };
    const float* bi[3] = { (const float*)d_in[4],  (const float*)d_in[8],  (const float*)d_in[12] };
    const float* ga[3] = { (const float*)d_in[5],  (const float*)d_in[9],  (const float*)d_in[13] };
    const float* be[3] = { (const float*)d_in[6],  (const float*)d_in[10], (const float*)d_in[14] };
    const float* fw1 = (const float*)d_in[15];
    const float* fb1 = (const float*)d_in[16];
    const float* fw2 = (const float*)d_in[17];
    const float* fb2 = (const float*)d_in[18];
    float* out = (float*)d_out;

    // workspace carve-up
    char* ws = (char*)d_ws;
    size_t off = 0;
    float* dis = (float*)(ws + off);          off = alignup(off + (size_t)N * 4);
    float* t   = (float*)(ws + off);          off = alignup(off + (size_t)N * H * 4);
    float* acc = (float*)(ws + off);          off = alignup(off + (size_t)N * H * 4);
    __bf16* hb = (__bf16*)(ws + off);         off = alignup(off + (size_t)N * H * 2);
    __bf16* Wb = (__bf16*)(ws + off);         off = alignup(off + (size_t)H * H * 2);
    float* pooled = (float*)(ws + off);       off = alignup(off + (size_t)B * H * 4);
    float* cnt    = (float*)(ws + off);       off = alignup(off + (size_t)B * 4);

    const long long NH = (long long)N * H;
    const int nb_elem = (int)((NH + 255) / 256);
    const int nb_node = (N + 7) / 8;          // wave-per-node kernels, 8 waves/blk
    const int nb_gemm = (N + 127) / 128;      // wave strip = 16 rows, 8 waves/blk

    // normalization: dis = rsqrt(1 + indeg)
    k_dis_init <<<(N + 255) / 256, 256, 0, stream>>>(dis, N);
    k_deg_edges<<<(E + 255) / 256, 256, 0, stream>>>(dis, ei, E);
    k_dis_rsqrt<<<(N + 255) / 256, 256, 0, stream>>>(dis, N);

    // bf16 input for the first WMMA GEMM
    k_to_bf16<<<nb_elem, 256, 0, stream>>>(x, hb, NH);

    for (int layer = 0; layer < 3; ++layer) {
        k_convW<<<(H * H + 255) / 256, 256, 0, stream>>>(W[layer], Wb);
        k_gemm_wmma<<<nb_gemm, 256, 0, stream>>>(hb, Wb, dis, t, N);
        k_copy<<<nb_elem, 256, 0, stream>>>(t, acc, NH);           // self-loops
        k_scatter<<<E / 8, 256, 0, stream>>>(t, acc, ei, E);       // edges
        k_finalize<<<nb_node, 256, 0, stream>>>(acc, dis, bi[layer], ga[layer],
                                                be[layer], t, hb, N);
    }

    k_zero<<<(B * H + B + 255) / 256, 256, 0, stream>>>(pooled, B * H + B); // pooled+cnt contiguous
    k_pool<<<nb_node, 256, 0, stream>>>(t, batch, pooled, cnt, N);
    k_mlp <<<B, 64, 0, stream>>>(pooled, cnt, fw1, fb1, fw2, fb2, out);
}
